// SimpleSpectralAttention_82910048682774
// MI455X (gfx1250) — matrix operand order repaired and checked by reading, not compiled, not run on hardware
//
#include <hip/hip_runtime.h>
#include <cstdint>

// Problem constants (match reference)
#define BB   2
#define LL   1024
#define DD   768
#define NHH  12
#define HDD  64
#define KF   24      // spectral filter count

typedef __attribute__((ext_vector_type(16))) __bf16 v16bf;
typedef __attribute__((ext_vector_type(8)))  float  v8f;

union Frag16 {           // 16 bf16 = 8 VGPRs (A or B fragment of 16x16x32 bf16 WMMA)
  uint4          u4[2];
  v16bf          v;
  unsigned short h[16];
};

__device__ __forceinline__ unsigned short f32_bf16(float f) {
  unsigned int x = __float_as_uint(f);
  x += 0x7FFFu + ((x >> 16) & 1u);          // round-to-nearest-even
  return (unsigned short)(x >> 16);
}

// CDNA5 async global -> LDS copy (16B per lane), tracked by ASYNCcnt
__device__ __forceinline__ void async_b128(unsigned lds_addr, const void* gptr) {
  asm volatile("global_load_async_to_lds_b128 %0, %1, off"
               :: "v"(lds_addr), "v"((unsigned long long)(uintptr_t)gptr)
               : "memory");
}

// ---------------------------------------------------------------------------
// elementwise f32 -> bf16
__global__ void __launch_bounds__(256)
cvt_bf16(const float* __restrict__ src, unsigned short* __restrict__ dst, int n) {
  int i = blockIdx.x * 256 + threadIdx.x;
  if (i < n) dst[i] = f32_bf16(src[i]);
}

// W (K x N, f32, row-major) -> WT (N x K, bf16)
__global__ void __launch_bounds__(256)
transpose_cvt(const float* __restrict__ W, unsigned short* __restrict__ WT,
              int K, int N) {
  int i = blockIdx.x * 256 + threadIdx.x;
  if (i >= K * N) return;
  int k = i / N, n = i - k * N;
  WT[(size_t)n * K + k] = f32_bf16(W[(size_t)k * N + n]);
}

// phi2[s][r] = (s even ? 2 : 0) * sum_k filters[s][k] * M_filters[k][r]
__global__ void __launch_bounds__(256)
phi_proj(const float* __restrict__ filt, const float* __restrict__ Mf,
         float* __restrict__ phi2) {
  int idx = blockIdx.x * 256 + threadIdx.x;
  if (idx >= LL * DD) return;
  int s = idx / DD, r = idx - s * DD;
  float acc = 0.f;
#pragma unroll
  for (int k = 0; k < KF; ++k) acc += filt[s * KF + k] * Mf[k * DD + r];
  phi2[idx] = (s & 1) ? 0.0f : 2.0f * acc;
}

// ---------------------------------------------------------------------------
// bf16 WMMA GEMM with async-LDS double-buffered staging.
//   C[M,N](f32 and/or bf16) = A[M,K] @ WT[N,K]^T + bias
// grid = (M/64, N/128), block = 256 (8 waves in 2x4; each wave owns 32x32 of C)
__global__ void __launch_bounds__(256)
gemm_bf16_wmma(const unsigned short* __restrict__ A,
               const unsigned short* __restrict__ WT,
               const float* __restrict__ bias,
               float* __restrict__ Cf,
               unsigned short* __restrict__ Ch,
               int M, int N, int K) {
  __shared__ unsigned short sA[2][64 * 32];    //  4 KB x2  (A tile, row-major MxK)
  __shared__ unsigned short sB[2][128 * 32];   //  8 KB x2  (WT tile, row-major NxK)

  const int tid  = threadIdx.x;
  const int wave = tid >> 5;
  const int lane = tid & 31;
  const int lm   = lane & 15;
  const int lh   = lane >> 4;
  const int wm   = (wave >> 2) * 32;           // wave's M offset in macro tile
  const int wn   = (wave & 3) * 32;            // wave's N offset in macro tile
  const int mtile = blockIdx.x * 64;
  const int ntile = blockIdx.y * 128;

  // per-thread staging coordinates: one b128 (8 bf16) per async op
  const int srow = tid >> 2;                   // 0..63
  const int scol = (tid & 3) * 8;              // 0,8,16,24

  const unsigned short* Ag = A  + (size_t)(mtile + srow) * K + scol;
  const unsigned short* Bg0 = WT + (size_t)(ntile + srow) * K + scol;
  const unsigned short* Bg1 = WT + (size_t)(ntile + srow + 64) * K + scol;

  v8f acc[2][2] = {};
  const int nk = K / 32;

  // prologue: stage k-chunk 0 into buffer 0 (3 async b128 per thread)
  async_b128((unsigned)(uintptr_t)&sA[0][srow * 32 + scol], Ag);
  async_b128((unsigned)(uintptr_t)&sB[0][srow * 32 + scol], Bg0);
  async_b128((unsigned)(uintptr_t)&sB[0][(srow + 64) * 32 + scol], Bg1);

  for (int i = 0; i < nk; ++i) {
    const int buf = i & 1;
    if (i + 1 < nk) {                          // stage next chunk into other buffer
      const int kc = (i + 1) * 32;
      async_b128((unsigned)(uintptr_t)&sA[buf ^ 1][srow * 32 + scol], Ag + kc);
      async_b128((unsigned)(uintptr_t)&sB[buf ^ 1][srow * 32 + scol], Bg0 + kc);
      async_b128((unsigned)(uintptr_t)&sB[buf ^ 1][(srow + 64) * 32 + scol], Bg1 + kc);
      asm volatile("s_wait_asynccnt 0x3" ::: "memory");   // current buffer landed
    } else {
      asm volatile("s_wait_asynccnt 0x0" ::: "memory");
    }
    __syncthreads();                           // all waves' staging visible

    Frag16 af[2], bfr[2];
#pragma unroll
    for (int ms = 0; ms < 2; ++ms) {           // A 16x32 frag: lane holds M=lm
      const unsigned short* ap = &sA[buf][(wm + ms * 16 + lm) * 32];
      af[ms].u4[0] = *(const uint4*)(ap + 8 * lh);
      af[ms].u4[1] = *(const uint4*)(ap + 16 + 8 * lh);
    }
#pragma unroll
    for (int ns = 0; ns < 2; ++ns) {           // B 32x16 frag: lane holds N=lm
      const unsigned short* bp = &sB[buf][(wn + ns * 16 + lm) * 32];
      bfr[ns].u4[0] = *(const uint4*)(bp + 8 * lh);
      bfr[ns].u4[1] = *(const uint4*)(bp + 16 + 8 * lh);
    }
#pragma unroll
    for (int ms = 0; ms < 2; ++ms)
#pragma unroll
      for (int ns = 0; ns < 2; ++ns)
        acc[ms][ns] = __builtin_amdgcn_wmma_f32_16x16x32_bf16(
            false, af[ms].v, false, bfr[ns].v, (short)0, acc[ms][ns], false, false);

    __syncthreads();                           // reads done before buf is re-staged
  }

#pragma unroll
  for (int ms = 0; ms < 2; ++ms)
#pragma unroll
    for (int ns = 0; ns < 2; ++ns) {
      const int   n  = ntile + wn + ns * 16 + lm;
      const float bv = bias ? bias[n] : 0.0f;
#pragma unroll
      for (int j = 0; j < 8; ++j) {            // C layout: M = j + 8*(lane>>4)
        const int   m   = mtile + wm + ms * 16 + j + 8 * lh;
        const float val = acc[ms][ns][j] + bv;
        if (Cf) Cf[(size_t)m * N + n] = val;
        if (Ch) Ch[(size_t)m * N + n] = f32_bf16(val);
      }
    }
}

// ---------------------------------------------------------------------------
// Per-channel causal spectral conv: xt[b,t,r] = sum_{s even<=t} phi2[s,r]*u[b,t-s,r]
__global__ void __launch_bounds__(256)
spectral_conv(const float* __restrict__ u, const float* __restrict__ phi2,
              float* __restrict__ xt) {
  const int r = blockIdx.x * 256 + threadIdx.x;
  const int t = blockIdx.y;
  const int b = blockIdx.z;
  const float* ub = u + (size_t)b * LL * DD + r;
  float acc = 0.f;
  for (int s = 0; s <= t; s += 2)
    acc += phi2[(size_t)s * DD + r] * ub[(size_t)(t - s) * DD];
  xt[((size_t)b * LL + t) * DD + r] = acc;
}

// ---------------------------------------------------------------------------
// Causal linear attention, one wave per (b, h, 16-row t-tile):
//   Y[t] = sum_{s<=t} (q[t].v[s]) k[s]
__global__ void __launch_bounds__(32)
linattn_wmma(const unsigned short* __restrict__ qh,
             const unsigned short* __restrict__ kh,
             const unsigned short* __restrict__ vh,
             float* __restrict__ yattn) {
  __shared__ float sS[16 * 17];                   // transpose scratch (pad=17)
  const int lane = threadIdx.x & 31;
  const int lm   = lane & 15;
  const int lh   = lane >> 4;
  const int koff = 8 * lh;

  const int ttile = blockIdx.x & 63;
  const int bh    = blockIdx.x >> 6;
  const int h     = bh % NHH;
  const int b     = bh / NHH;
  const size_t base = (size_t)b * LL * DD + (size_t)h * HDD;

  // Q fragment (16 rows x 64 head-dim), held in registers across the s loop
  const unsigned short* qrow = qh + base + (size_t)(ttile * 16 + lm) * DD;
  Frag16 qf[2];
#pragma unroll
  for (int kk = 0; kk < 2; ++kk) {
    qf[kk].u4[0] = *(const uint4*)(qrow + kk * 32 + koff);
    qf[kk].u4[1] = *(const uint4*)(qrow + kk * 32 + 16 + koff);
  }

  v8f yacc[4] = {};

  for (int st = 0; st <= ttile; ++st) {
    const int s0 = st * 16;

    // S = Q @ V^T  (16x16 f32, K = 64 as two x32 steps)
    v8f sacc = {};
    const unsigned short* vrow = vh + base + (size_t)(s0 + lm) * DD;
#pragma unroll
    for (int kk = 0; kk < 2; ++kk) {
      Frag16 vb;
      vb.u4[0] = *(const uint4*)(vrow + kk * 32 + koff);
      vb.u4[1] = *(const uint4*)(vrow + kk * 32 + 16 + koff);
      sacc = __builtin_amdgcn_wmma_f32_16x16x32_bf16(false, qf[kk].v, false, vb.v,
                                                     (short)0, sacc, false, false);
    }

    // causal mask on diagonal tile, stash S[t_local][s_local] in LDS
#pragma unroll
    for (int j = 0; j < 8; ++j) {
      const int tl = j + 8 * lh;
      float v = sacc[j];
      if (st == ttile && lm > tl) v = 0.0f;
      sS[tl * 17 + lm] = v;
    }
    __syncthreads();                              // single wave -> waits only

    // rebuild S as bf16 A-fragment (K = 16 real, upper 16 zero-padded)
    Frag16 sa;
#pragma unroll
    for (int i = 0; i < 16; ++i) sa.h[i] = lh ? (unsigned short)0 : f32_bf16(sS[lm * 17 + i]);
    __syncthreads();

    // Y += S @ K  (16x64, four 16-wide N tiles)
#pragma unroll
    for (int nt = 0; nt < 4; ++nt) {
      Frag16 kb;
      if (lh == 0) {                              // K rows 0..15 live in low lanes
        const unsigned short* kcol = kh + base + (size_t)s0 * DD + nt * 16 + lm;
#pragma unroll
        for (int sl = 0; sl < 16; ++sl) kb.h[sl] = kcol[(size_t)sl * DD];
      } else {                                    // zero pad K = 16..31
#pragma unroll
        for (int sl = 0; sl < 16; ++sl) kb.h[sl] = 0;
      }
      yacc[nt] = __builtin_amdgcn_wmma_f32_16x16x32_bf16(false, sa.v, false, kb.v,
                                                         (short)0, yacc[nt], false, false);
    }
  }

#pragma unroll
  for (int nt = 0; nt < 4; ++nt)
#pragma unroll
    for (int j = 0; j < 8; ++j) {
      const int t = ttile * 16 + j + 8 * lh;
      yattn[base + (size_t)t * DD + nt * 16 + lm] = yacc[nt][j];
    }
}

// ---------------------------------------------------------------------------
// y = x_tilde + (y_attn - x_tilde) * sigmoid(g); emit bf16 for the output GEMM
__global__ void __launch_bounds__(256)
gate_combine(const float* __restrict__ xt, const float* __restrict__ ya,
             const float* __restrict__ g, unsigned short* __restrict__ yb, int n) {
  int i = blockIdx.x * 256 + threadIdx.x;
  if (i >= n) return;
  float gate = 1.0f / (1.0f + __expf(-g[i]));
  float y    = xt[i] + (ya[i] - xt[i]) * gate;
  yb[i] = f32_bf16(y);
}

// ---------------------------------------------------------------------------
extern "C" void kernel_launch(void* const* d_in, const int* in_sizes, int n_in,
                              void* d_out, int out_size, void* d_ws, size_t ws_size,
                              hipStream_t stream) {
  const float* x    = (const float*)d_in[0];
  const float* filt = (const float*)d_in[1];   // (L, 24)
  const float* Mi   = (const float*)d_in[2];   // (768, 768)
  const float* Mf   = (const float*)d_in[3];   // (24, 768)
  const float* wq   = (const float*)d_in[4];  const float* bq = (const float*)d_in[5];
  const float* wk   = (const float*)d_in[6];  const float* bk = (const float*)d_in[7];
  const float* wv   = (const float*)d_in[8];  const float* bv = (const float*)d_in[9];
  const float* wg   = (const float*)d_in[10]; const float* bg = (const float*)d_in[11];
  const float* wo   = (const float*)d_in[12]; const float* bo = (const float*)d_in[13];
  float* out = (float*)d_out;

  const int    MROWS = BB * LL;           // 2048
  const size_t NX    = (size_t)MROWS * DD;
  const size_t NW    = (size_t)DD * DD;

  char* p = (char*)d_ws;
  auto carve = [&](size_t bytes) { char* r = p; p += (bytes + 255) & ~(size_t)255; return r; };
  unsigned short* xb   = (unsigned short*)carve(NX * 2);
  unsigned short* miT  = (unsigned short*)carve(NW * 2);
  unsigned short* wqT  = (unsigned short*)carve(NW * 2);
  unsigned short* wkT  = (unsigned short*)carve(NW * 2);
  unsigned short* wvT  = (unsigned short*)carve(NW * 2);
  unsigned short* wgT  = (unsigned short*)carve(NW * 2);
  unsigned short* woT  = (unsigned short*)carve(NW * 2);
  float*          u    = (float*)carve(NX * 4);
  float*          phi2 = (float*)carve((size_t)LL * DD * 4);
  float*          xt   = (float*)carve(NX * 4);
  unsigned short* qhb  = (unsigned short*)carve(NX * 2);
  unsigned short* khb  = (unsigned short*)carve(NX * 2);
  unsigned short* vhb  = (unsigned short*)carve(NX * 2);
  float*          gb   = (float*)carve(NX * 4);
  float*          ya   = (float*)carve(NX * 4);
  unsigned short* yb   = (unsigned short*)carve(NX * 2);

  // 1) precision conversion (one-time; traffic negligible vs. L2-resident compute)
  cvt_bf16<<<(int)((NX + 255) / 256), 256, 0, stream>>>(x, xb, (int)NX);
  const int trb = (int)((NW + 255) / 256);
  transpose_cvt<<<trb, 256, 0, stream>>>(Mi, miT, DD, DD);
  transpose_cvt<<<trb, 256, 0, stream>>>(wq, wqT, DD, DD);
  transpose_cvt<<<trb, 256, 0, stream>>>(wk, wkT, DD, DD);
  transpose_cvt<<<trb, 256, 0, stream>>>(wv, wvT, DD, DD);
  transpose_cvt<<<trb, 256, 0, stream>>>(wg, wgT, DD, DD);
  transpose_cvt<<<trb, 256, 0, stream>>>(wo, woT, DD, DD);

  // 2) phi2 = 2 * even-rows of (stu_filters @ M_filters)
  phi_proj<<<(LL * DD) / 256, 256, 0, stream>>>(filt, Mf, phi2);

  // 3) WMMA GEMMs (async-LDS double buffered): u and the four projections
  dim3 ggrid(MROWS / 64, DD / 128);       // (32, 6)
  gemm_bf16_wmma<<<ggrid, 256, 0, stream>>>(xb, miT, nullptr, u, nullptr, MROWS, DD, DD);
  gemm_bf16_wmma<<<ggrid, 256, 0, stream>>>(xb, wqT, bq, nullptr, qhb, MROWS, DD, DD);
  gemm_bf16_wmma<<<ggrid, 256, 0, stream>>>(xb, wkT, bk, nullptr, khb, MROWS, DD, DD);
  gemm_bf16_wmma<<<ggrid, 256, 0, stream>>>(xb, wvT, bv, nullptr, vhb, MROWS, DD, DD);
  gemm_bf16_wmma<<<ggrid, 256, 0, stream>>>(xb, wgT, bg, gb, nullptr, MROWS, DD, DD);

  // 4) spectral causal conv (per-channel Toeplitz, f32 VALU, L2-resident)
  spectral_conv<<<dim3(DD / 256, LL, BB), 256, 0, stream>>>(u, phi2, xt);

  // 5) causal linear attention via tiled WMMA
  linattn_wmma<<<BB * NHH * (LL / 16), 32, 0, stream>>>(qhb, khb, vhb, ya);

  // 6) gated mix, then output projection GEMM straight into d_out
  gate_combine<<<(int)((NX + 255) / 256), 256, 0, stream>>>(xt, ya, gb, yb, (int)NX);
  gemm_bf16_wmma<<<ggrid, 256, 0, stream>>>(yb, woT, bo, out, nullptr, MROWS, DD, DD);
}
